// EmbLearnerWithoutHyper_74749610820268
// MI455X (gfx1250) — matrix operand order repaired and checked by reading, not compiled, not run on hardware
//
#include <hip/hip_runtime.h>
#include <hip/hip_bf16.h>
#include <cstdint>
#include <cstddef>

// ---------------------------------------------------------------------------
// Types for WMMA (gfx1250, wave32)
// ---------------------------------------------------------------------------
typedef __attribute__((ext_vector_type(16))) __bf16        v16bf;
typedef __attribute__((ext_vector_type(8)))  float         v8f;
typedef __attribute__((ext_vector_type(4)))  unsigned int  v4u;
typedef __attribute__((ext_vector_type(8)))  unsigned int  v8u;

__device__ __forceinline__ unsigned short f2bf(float f) {
    unsigned u = __builtin_bit_cast(unsigned, f);
    u += 0x7FFFu + ((u >> 16) & 1u);          // round-to-nearest-even
    return (unsigned short)(u >> 16);
}

// ---------------------------------------------------------------------------
// Degree / normalization kernels
// ---------------------------------------------------------------------------
__global__ __launch_bounds__(256) void fill1_kernel(float* deg, int n) {
    int i = blockIdx.x * 256 + threadIdx.x;
    if (i < n) deg[i] = 1.0f;                 // self-loop contribution (+1)
}

__global__ __launch_bounds__(256) void deg_scatter_kernel(const int* __restrict__ dst,
                                                          float* deg, int E) {
    int e = blockIdx.x * 256 + threadIdx.x;
    if (e < E) atomicAdd(&deg[dst[e]], 1.0f);
}

__global__ __launch_bounds__(256) void deg_fin_kernel(const float* __restrict__ deg,
                                                      float* dinv, float* snorm, int n) {
    int i = blockIdx.x * 256 + threadIdx.x;
    if (i < n) {
        float d = deg[i];
        dinv[i]  = rsqrtf(d);
        snorm[i] = 1.0f / d;
    }
}

__global__ __launch_bounds__(256) void enorm_kernel(const int* __restrict__ src,
                                                    const int* __restrict__ dst,
                                                    const float* __restrict__ dinv,
                                                    float* enorm, int E) {
    int e = blockIdx.x * 256 + threadIdx.x;
    if (e < E) enorm[e] = dinv[src[e]] * dinv[dst[e]];
}

// ---------------------------------------------------------------------------
// f32 -> bf16 conversion (row-major activations)
// ---------------------------------------------------------------------------
__global__ __launch_bounds__(256) void to_bf16_kernel(const float* __restrict__ in,
                                                      unsigned short* __restrict__ out,
                                                      size_t total) {
    size_t i = (size_t)blockIdx.x * 256 + threadIdx.x;
    if (i < total) out[i] = f2bf(in[i]);
}

// ---------------------------------------------------------------------------
// Weight swizzle: f32 [K,Hc] row-major -> bf16 in WMMA B-fragment order
// layout: [kb][ng][lane][j], j in 0..15 contiguous per lane.
//   col = ng*16 + (lane&15)
//   kk  = kb*32 + (j&7) + (j>>3)*16 + (lane>>4)*8     (16-bit B 32x16 lane layout)
// ---------------------------------------------------------------------------
__global__ __launch_bounds__(256) void swizzle_kernel(const float* __restrict__ W,
                                                      unsigned short* __restrict__ out,
                                                      int K, int Hc) {
    size_t t = (size_t)blockIdx.x * 256 + threadIdx.x;
    if (t >= (size_t)K * Hc) return;
    int j      = (int)(t & 15);
    int lane   = (int)((t >> 4) & 31);
    size_t rest = t >> 9;
    int groups = Hc >> 4;
    int ng = (int)(rest % groups);
    int kb = (int)(rest / groups);
    int kk  = kb * 32 + (j & 7) + (j >> 3) * 16 + (lane >> 4) * 8;
    int col = ng * 16 + (lane & 15);
    out[t] = f2bf(W[(size_t)kk * Hc + col]);
}

// ---------------------------------------------------------------------------
// GEMM: C[M,Hc] = A_bf16[M,K] @ B_swizzled_bf16[K,Hc] (+bias) (+relu)
// Block 256 = 8 waves; tile 128x128; wave tile 32x64 (2x4 x 16x16 frags).
// ---------------------------------------------------------------------------
__global__ __launch_bounds__(256) void gemm_bf16_wmma(
        const unsigned short* __restrict__ A,   // [M,K] bf16 row-major
        const unsigned short* __restrict__ Bs,  // swizzled weight
        float* __restrict__ Cf,                 // may be null
        unsigned short* __restrict__ Cb,        // may be null (bf16 copy)
        const float* __restrict__ bias,         // may be null
        int M, int K, int Hc, int relu)
{
    const int tid   = threadIdx.x;
    const int lane  = tid & 31;
    const int wave  = tid >> 5;
    const int waveM = wave & 3;                 // 4 waves along M
    const int waveN = wave >> 2;                // 2 waves along N
    const int rbase = blockIdx.x * 128 + waveM * 32;
    const int cbase = blockIdx.y * 128 + waveN * 64;
    const int l16   = lane & 15;
    const int lh8   = (lane >> 4) * 8;          // A K-offset per lane-half
    const int groups = Hc >> 4;

    v8f acc[2][4] = {};

    const int kb_count = K >> 5;
    for (int kb = 0; kb < kb_count; ++kb) {
        const int k0 = kb << 5;
        v16bf afrag[2];
        #pragma unroll
        for (int fm = 0; fm < 2; ++fm) {
            int row = rbase + fm * 16 + l16;
            row = row < M ? row : M - 1;        // clamp (dead rows masked at store)
            const unsigned short* ap = A + (size_t)row * K + k0 + lh8;
            union { v8u v; v4u h[2]; } au;
            au.h[0] = *(const v4u*)(ap);        // K: +0..7   (lane<16) / +8..15
            au.h[1] = *(const v4u*)(ap + 16);   // K: +16..23 (lane<16) / +24..31
            afrag[fm] = __builtin_bit_cast(v16bf, au.v);
        }
        #pragma unroll
        for (int ng = 0; ng < 4; ++ng) {
            const int ngi = (cbase >> 4) + ng;
            const unsigned short* bp =
                Bs + ((((size_t)kb * groups + ngi) * 32 + lane) << 4);
            union { v8u v; v4u h[2]; } bu;
            bu.h[0] = *(const v4u*)(bp);
            bu.h[1] = *(const v4u*)(bp + 8);
            v16bf bfrag = __builtin_bit_cast(v16bf, bu.v);
            acc[0][ng] = __builtin_amdgcn_wmma_f32_16x16x32_bf16(
                false, afrag[0], false, bfrag, (short)0, acc[0][ng], false, false);
            acc[1][ng] = __builtin_amdgcn_wmma_f32_16x16x32_bf16(
                false, afrag[1], false, bfrag, (short)0, acc[1][ng], false, false);
        }
    }

    // Epilogue. C/D layout: VGPR v -> row = v + (lane>=16 ? 8 : 0); col = lane&15.
    #pragma unroll
    for (int fm = 0; fm < 2; ++fm) {
        #pragma unroll
        for (int ng = 0; ng < 4; ++ng) {
            int col = cbase + ng * 16 + l16;
            float bv = bias ? bias[col] : 0.0f;
            #pragma unroll
            for (int v = 0; v < 8; ++v) {
                int row = rbase + fm * 16 + v + (lh8 & 8);
                if (row < M) {
                    float val = acc[fm][ng][v] + bv;
                    if (relu) val = fmaxf(val, 0.0f);
                    size_t o = (size_t)row * Hc + col;
                    if (Cf) Cf[o] = val;
                    if (Cb) Cb[o] = f2bf(val);
                }
            }
        }
    }
}

// ---------------------------------------------------------------------------
// gcn init: out = snorm[n]*xw + bias[h] (+ extra)  (float4 vectorized)
// ---------------------------------------------------------------------------
__global__ __launch_bounds__(256) void gcn_init_kernel(const float* __restrict__ xw,
                                                       const float* __restrict__ snorm,
                                                       const float* __restrict__ bias,
                                                       const float* __restrict__ extra,
                                                       float* __restrict__ out,
                                                       int N, int H) {
    size_t i = (size_t)blockIdx.x * 256 + threadIdx.x;
    size_t total = ((size_t)N * H) >> 2;
    if (i >= total) return;
    size_t base = i << 2;
    int n = (int)(base / H);
    int h = (int)(base % H);
    float s = snorm[n];
    float4 x = *(const float4*)(xw + base);
    float4 b = *(const float4*)(bias + h);
    float4 r = make_float4(0.f, 0.f, 0.f, 0.f);
    if (extra) r = *(const float4*)(extra + base);
    float4 o;
    o.x = s * x.x + b.x + r.x;
    o.y = s * x.y + b.y + r.y;
    o.z = s * x.z + b.z + r.z;
    o.w = s * x.w + b.w + r.w;
    *(float4*)(out + base) = o;
}

// ---------------------------------------------------------------------------
// Edge scatter: out[dst] += enorm[e] * xw[src]   (64 threads/edge, float4)
// ---------------------------------------------------------------------------
__global__ __launch_bounds__(256) void edge_scatter_kernel(const int* __restrict__ src,
                                                           const int* __restrict__ dst,
                                                           const float* __restrict__ enorm,
                                                           const float* __restrict__ xw,
                                                           float* __restrict__ out,
                                                           int E, int H) {
    size_t t = (size_t)blockIdx.x * 256 + threadIdx.x;
    size_t e = t >> 6;
    if (e >= (size_t)E) return;
    int h = (int)(t & 63) << 2;
    float w = enorm[e];
    int s = src[e], d = dst[e];
    float4 x = *(const float4*)(xw + (size_t)s * H + h);
    float* op = out + (size_t)d * H + h;
    atomicAdd(op + 0, w * x.x);
    atomicAdd(op + 1, w * x.y);
    atomicAdd(op + 2, w * x.z);
    atomicAdd(op + 3, w * x.w);
}

// ---------------------------------------------------------------------------
// Post: optional in-place relu, optional bf16 copy
// ---------------------------------------------------------------------------
__global__ __launch_bounds__(256) void post_kernel(float* __restrict__ io,
                                                   unsigned short* __restrict__ ob,
                                                   int relu, size_t total) {
    size_t i = (size_t)blockIdx.x * 256 + threadIdx.x;
    if (i >= total) return;
    float v = io[i];
    if (relu) { v = fmaxf(v, 0.0f); io[i] = v; }
    if (ob) ob[i] = f2bf(v);
}

// ---------------------------------------------------------------------------
// One-hot row: out[n,h] = (bias?bias[h]:0) + (n==*qidx ? row[h] : 0)
// ---------------------------------------------------------------------------
__global__ __launch_bounds__(256) void set_row_kernel(float* __restrict__ out,
                                                      const float* __restrict__ row,
                                                      const float* __restrict__ bias,
                                                      const int* __restrict__ qidx,
                                                      int N, int H) {
    size_t i = (size_t)blockIdx.x * 256 + threadIdx.x;
    if (i >= (size_t)N * H) return;
    int n = (int)(i / H);
    int h = (int)(i % H);
    float v = bias ? bias[h] : 0.0f;
    if (n == *qidx) v += row[h];
    out[i] = v;
}

// ---------------------------------------------------------------------------
// Fuse: 2-way softmax attention.  One block (H threads) per node.
// ---------------------------------------------------------------------------
__global__ __launch_bounds__(256) void fuse_kernel(const float* __restrict__ hq,
                                                   const float* __restrict__ h,
                                                   const float* __restrict__ aq,
                                                   const float* __restrict__ ah,
                                                   float* __restrict__ out,
                                                   unsigned short* __restrict__ ob,
                                                   int H) {
    __shared__ float s0[256];
    __shared__ float s1[256];
    int n = blockIdx.x;
    int t = threadIdx.x;
    size_t idx = (size_t)n * H + t;
    float x = hq[idx], y = h[idx];
    s0[t] = x * aq[t];
    s1[t] = y * ah[t];
    __syncthreads();
    for (int off = blockDim.x >> 1; off > 0; off >>= 1) {
        if (t < off) { s0[t] += s0[t + off]; s1[t] += s1[t + off]; }
        __syncthreads();
    }
    float a = s0[0], b = s1[0];
    float m  = fmaxf(a, b);
    float e0 = __expf(a - m), e1 = __expf(b - m);
    float inv = 1.0f / (e0 + e1);
    float o = (e0 * x + e1 * y) * inv;
    out[idx] = o;
    if (ob) ob[idx] = f2bf(o);
}

// ---------------------------------------------------------------------------
// Host orchestration
// ---------------------------------------------------------------------------
extern "C" void kernel_launch(void* const* d_in, const int* in_sizes, int n_in,
                              void* d_out, int out_size, void* d_ws, size_t ws_size,
                              hipStream_t stream) {
    (void)n_in; (void)out_size; (void)ws_size;

    const int H = in_sizes[2];            // 256
    const int F = in_sizes[5] / H;        // 128
    const int N = in_sizes[0] / F;        // 20000
    const int E = in_sizes[24] / 2;       // 320000
    const size_t NH = (size_t)N * H;
    const size_t HH = (size_t)H * H;

    const float* feats  = (const float*)d_in[0];
    const float* qW0    = (const float*)d_in[1];
    const float* qb0    = (const float*)d_in[2];
    const float* qW     = (const float*)d_in[3];
    const float* qb     = (const float*)d_in[4];
    const float* xW0    = (const float*)d_in[5];
    const float* xb0    = (const float*)d_in[6];
    const float* xW     = (const float*)d_in[7];
    const float* xb     = (const float*)d_in[8];
    const float* fW     = (const float*)d_in[9];
    const float* fb     = (const float*)d_in[10];
    const float* qatt   = (const float*)d_in[11];
    const float* att    = (const float*)d_in[12];
    const float* fq_att = (const float*)d_in[13];
    const float* f_att  = (const float*)d_in[14];
    const float* lqW    = (const float*)d_in[15];
    const float* lqb    = (const float*)d_in[16];
    const float* lfW    = (const float*)d_in[17];
    const float* lfb    = (const float*)d_in[18];
    const float* m1W    = (const float*)d_in[19];
    const float* m1b    = (const float*)d_in[20];
    const float* m2W    = (const float*)d_in[21];
    const float* m2b    = (const float*)d_in[22];
    const int*   qidx   = (const int*)d_in[23];
    const int*   ei     = (const int*)d_in[24];
    const int*   src    = ei;
    const int*   dst    = ei + E;

    uintptr_t wp = (uintptr_t)d_ws;
    auto alloc = [&](size_t bytes) -> void* {
        wp = (wp + 255) & ~(uintptr_t)255;
        void* p = (void*)wp;
        wp += bytes;
        return p;
    };

    float* deg   = (float*)alloc((size_t)N * 4);
    float* dinv  = (float*)alloc((size_t)N * 4);
    float* snorm = (float*)alloc((size_t)N * 4);
    float* enorm = (float*)alloc((size_t)E * 4);
    float* B0 = (float*)alloc(NH * 4);   // GEMM scratch (xw)
    float* B1 = (float*)alloc(NH * 4);   // hq
    float* B2 = (float*)alloc(NH * 4);   // h
    float* B3 = (float*)alloc(NH * 4);   // hf / hfx
    float* B4 = (float*)alloc(NH * 4);   // flin / hf
    float* B5 = (float*)alloc(NH * 4);   // qlin / hf_ / hf
    unsigned short* featsb = (unsigned short*)alloc((size_t)N * F * 2);
    unsigned short* bfA = (unsigned short*)alloc(NH * 2);   // hq_bf / t1_bf
    unsigned short* bfB = (unsigned short*)alloc(NH * 2);   // h_bf
    unsigned short* bfC = (unsigned short*)alloc(NH * 2);   // hf_/hf bf
    unsigned short* s_xW0 = (unsigned short*)alloc((size_t)F * H * 2);
    unsigned short* s_lfW = (unsigned short*)alloc((size_t)F * H * 2);
    unsigned short* s_qWa = (unsigned short*)alloc(HH * 2);
    unsigned short* s_qWb = (unsigned short*)alloc(HH * 2);
    unsigned short* s_xWa = (unsigned short*)alloc(HH * 2);
    unsigned short* s_xWb = (unsigned short*)alloc(HH * 2);
    unsigned short* s_fW0 = (unsigned short*)alloc(HH * 2);
    unsigned short* s_fW1 = (unsigned short*)alloc(HH * 2);
    unsigned short* s_fW2 = (unsigned short*)alloc(HH * 2);
    unsigned short* s_m1  = (unsigned short*)alloc(HH * 2);
    unsigned short* s_m2  = (unsigned short*)alloc(HH * 2);

    auto cdiv = [](size_t a, size_t b) -> unsigned { return (unsigned)((a + b - 1) / b); };

    // --- degree / normalization ---
    fill1_kernel<<<cdiv(N, 256), 256, 0, stream>>>(deg, N);
    deg_scatter_kernel<<<cdiv(E, 256), 256, 0, stream>>>(dst, deg, E);
    deg_fin_kernel<<<cdiv(N, 256), 256, 0, stream>>>(deg, dinv, snorm, N);
    enorm_kernel<<<cdiv(E, 256), 256, 0, stream>>>(src, dst, dinv, enorm, E);

    // --- bf16 feats + swizzled weights ---
    to_bf16_kernel<<<cdiv((size_t)N * F, 256), 256, 0, stream>>>(feats, featsb, (size_t)N * F);
    auto swz = [&](const float* W, unsigned short* o, int K) {
        swizzle_kernel<<<cdiv((size_t)K * H, 256), 256, 0, stream>>>(W, o, K, H);
    };
    swz(xW0, s_xW0, F);          swz(lfW, s_lfW, F);
    swz(qW, s_qWa, H);           swz(qW + HH, s_qWb, H);
    swz(xW, s_xWa, H);           swz(xW + HH, s_xWb, H);
    swz(fW, s_fW0, H);           swz(fW + HH, s_fW1, H);   swz(fW + 2 * HH, s_fW2, H);
    swz(m1W, s_m1, H);           swz(m2W, s_m2, H);

    auto gemm = [&](const unsigned short* Ain, const unsigned short* Ws, int K,
                    float* Cf, unsigned short* Cb, const float* bias, int relu) {
        dim3 grid(cdiv(N, 128), H / 128);
        gemm_bf16_wmma<<<grid, 256, 0, stream>>>(Ain, Ws, Cf, Cb, bias, N, K, H, relu);
    };
    auto gcn_agg = [&](const float* xw, const float* bias, const float* extra,
                       float* out, unsigned short* ob, int relu) {
        gcn_init_kernel<<<cdiv(NH / 4, 256), 256, 0, stream>>>(xw, snorm, bias, extra, out, N, H);
        edge_scatter_kernel<<<cdiv((size_t)E * 64, 256), 256, 0, stream>>>(src, dst, enorm, xw, out, E, H);
        if (relu || ob)
            post_kernel<<<cdiv(NH, 256), 256, 0, stream>>>(out, ob, relu, NH);
    };
    auto fuse = [&](const float* a, const float* b, const float* aq, const float* ah,
                    float* out, unsigned short* ob) {
        fuse_kernel<<<N, H, 0, stream>>>(a, b, aq, ah, out, ob, H);
    };

    // --- layer 0 ---
    set_row_kernel<<<cdiv(NH, 256), 256, 0, stream>>>(B0, qW0, nullptr, qidx, N, H);
    gcn_agg(B0, qb0, nullptr, B1, bfA, 1);                 // hq = relu(gcn(querys,qW0))
    gemm(featsb, s_xW0, F, B0, nullptr, nullptr, 0);
    gcn_agg(B0, xb0, nullptr, B2, bfB, 1);                 // h  = relu(gcn(feats,xW0))
    fuse(B1, B2, qatt, att, B3, nullptr);                  // hf
    set_row_kernel<<<cdiv(NH, 256), 256, 0, stream>>>(B5, lqW, lqb, qidx, N, H);  // qlin
    gemm(featsb, s_lfW, F, B4, nullptr, lfb, 0);           // flin
    fuse(B5, B4, fq_att, f_att, B5, bfC);                  // hf_  (f32 B5, bf16 bfC)
    gemm(bfC, s_fW0, H, B0, nullptr, nullptr, 0);
    gcn_agg(B0, fb, B3, B4, bfC, 1);                       // hf = relu(hf + gcn(hf_,fW0))

    // --- middle layer ---
    gemm(bfA, s_qWa, H, B0, nullptr, nullptr, 0);
    gcn_agg(B0, qb, nullptr, B1, bfA, 1);                  // hq
    gemm(bfB, s_xWa, H, B0, nullptr, nullptr, 0);
    gcn_agg(B0, xb, nullptr, B2, bfB, 1);                  // h
    fuse(B1, B2, qatt + H, att + H, B3, nullptr);          // hfx
    gemm(bfC, s_fW1, H, B0, nullptr, nullptr, 0);
    gcn_agg(B0, fb + H, B3, B5, bfC, 1);                   // hf = relu(hfx + gcn(hf,fW1))

    // --- last layer ---
    gemm(bfA, s_qWb, H, B0, nullptr, nullptr, 0);
    gcn_agg(B0, qb + H, nullptr, B1, nullptr, 0);          // hq (no relu)
    gemm(bfB, s_xWb, H, B0, nullptr, nullptr, 0);
    gcn_agg(B0, xb + H, nullptr, B2, nullptr, 0);          // h  (no relu)
    fuse(B1, B2, qatt + 2 * H, att + 2 * H, B3, nullptr);  // hfx
    gemm(bfC, s_fW2, H, B0, nullptr, nullptr, 0);
    gcn_agg(B0, fb + 2 * H, B3, B4, bfC, 0);               // hf = hfx + gcn(hf,fW2) ; bf16 -> bfC

    // --- projection MLP ---
    gemm(bfC, s_m1, H, nullptr, bfA, m1b, 1);              // t1 = relu(hf@m1W+m1b) (bf16 only)
    gemm(bfA, s_m2, H, (float*)d_out, nullptr, m2b, 0);    // out = t1@m2W + m2b
}
